// StructuredAttentionNextVLADPriorHead_6511170420904
// MI455X (gfx1250) — matrix operand order repaired and checked by reading, not compiled, not run on hardware
//
#include <hip/hip_runtime.h>
#include <hip/hip_bf16.h>

// ---------------------------------------------------------------------------
// Problem constants (from the reference)
// ---------------------------------------------------------------------------
constexpr int Bc  = 32;
constexpr int Tc  = 512;
constexpr int Dc  = 1024;
constexpr int Hc  = 8;
constexpr int DHc = 128;
constexpr int Ec  = 2048;
constexpr int Gc  = 8;
constexpr int DGc = 256;
constexpr int Kc  = 32;
constexpr int HIDc = 2048;
constexpr int NBc = 4;
constexpr int NCc = 3862;
constexpr int BT  = Bc * Tc;   // 16384

// ---------------------------------------------------------------------------
// Types / helpers
// ---------------------------------------------------------------------------
typedef __bf16 bf16_t;
typedef __attribute__((ext_vector_type(16))) __bf16 v16bf;
typedef __attribute__((ext_vector_type(8)))  float  v8f;

union FragBF { v16bf v; uint4 q[2]; };

__device__ __forceinline__ float gelu_f(float x) {
  // exact gelu: 0.5*x*(1+erf(x/sqrt(2)))
  return 0.5f * x * (1.0f + erff(x * 0.70710678118654752f));
}

__device__ __forceinline__ bf16_t f2bf(float f) {
  // round-to-nearest-even fp32 -> bf16
  unsigned int u = __builtin_bit_cast(unsigned int, f);
  u += 0x7FFFu + ((u >> 16) & 1u);
  unsigned short hs = (unsigned short)(u >> 16);
  return __builtin_bit_cast(bf16_t, hs);
}

// ---------------------------------------------------------------------------
// Generic WMMA GEMM:  C = epilogue( op(A) @ W + bias )
//   op(A) = A [M,K] row-major (transA=0)  or  A^T with A stored [K,M] (transA=1)
//   epilogue: act==1 -> GELU ;  rowscale (per-row multiplier) ;  addsrc (residual)
//   batched via blockIdx.z with element strides sA/sW/sC.
//
// Block: 128 threads (4 waves). Tile: (64*MFRAG)(M) x 64(N), K-step 32.
// Each wave computes a (16*MFRAG)x64 strip = 4*MFRAG v_wmma_f32_16x16x32_bf16
// per K-step. All fragments are preloaded before the WMMA burst so a single
// s_wait_dscnt covers the whole MFMA-style issue group.
// ---------------------------------------------------------------------------
#define TN 64
#define TKK 32
#define GEMM_THREADS 128

template <int MFRAG>   // 1 -> 64-row tile, 2 -> 128-row tile
__global__ __launch_bounds__(GEMM_THREADS)
void gemm_wmma_kernel(const float* __restrict__ A, const float* __restrict__ W,
                      const float* __restrict__ bias, const float* __restrict__ rowscale,
                      const float* __restrict__ addsrc, float* __restrict__ C,
                      int M, int N, int Kd, int lda, int ldw, int ldc,
                      long sA, long sW, long sC, int transA, int act) {
  constexpr int TMt = 64 * MFRAG;       // block M tile
  constexpr int WR  = 16 * MFRAG;       // rows per wave

  const int z = blockIdx.z;
  A += (long)z * sA;
  W += (long)z * sW;
  C += (long)z * sC;
  if (addsrc) addsrc += (long)z * sC;

  const int tileM = blockIdx.y * TMt;
  const int tileN = blockIdx.x * TN;

  // padded stride: 40 bf16 = 80 bytes -> rows stay 16B aligned, banks spread
  __shared__ __align__(16) bf16_t As[TMt][TKK + 8];
  __shared__ __align__(16) bf16_t Bs[TN][TKK + 8];   // stored transposed: Bs[n][k]

  const int tid  = threadIdx.x;
  const int wave = tid >> 5;
  const int lane = tid & 31;
  const int hf   = lane >> 4;     // which half of the wave
  const int nl   = lane & 15;

  v8f acc[MFRAG][4];
#pragma unroll
  for (int i = 0; i < MFRAG; ++i)
#pragma unroll
    for (int c = 0; c < 4; ++c)
#pragma unroll
      for (int r = 0; r < 8; ++r) acc[i][c][r] = 0.0f;

  for (int k0 = 0; k0 < Kd; k0 += TKK) {
    // ---- speculative prefetch of next K tile (global_prefetch_b8) ----
    if (k0 + TKK < Kd) {
      int pm = tileM + (tid % TMt);
      if (!transA && pm < M) __builtin_prefetch(&A[(long)pm * lda + k0 + TKK], 0, 1);
      int pk = k0 + TKK + (tid >> 6);
      if (pk < Kd) __builtin_prefetch(&W[(long)pk * ldw + tileN + (tid & 63)], 0, 1);
    }

    // ---- stage A tile (TMt x 32) as bf16 ----
    for (int idx = tid; idx < TMt * TKK; idx += GEMM_THREADS) {
      const int m = idx >> 5;
      const int k = idx & 31;
      const int gm = tileM + m, gk = k0 + k;
      float v = 0.0f;
      if (gm < M && gk < Kd)
        v = transA ? A[(long)gk * lda + gm] : A[(long)gm * lda + gk];
      As[m][k] = f2bf(v);
    }
    // ---- stage B tile (32 x 64) transposed as bf16: Bs[n][k] ----
    for (int idx = tid; idx < TN * TKK; idx += GEMM_THREADS) {
      const int k = idx >> 6;
      const int n = idx & 63;
      const int gk = k0 + k, gn = tileN + n;
      float v = 0.0f;
      if (gk < Kd && gn < N) v = W[(long)gk * ldw + gn];
      Bs[n][k] = f2bf(v);
    }
    __syncthreads();

    // ---- preload ALL fragments, then one WMMA burst ----
    // A fragment: lane nl -> row; halves hold K 0-7/8-15 then 16-23/24-31
    FragBF fa[MFRAG];
#pragma unroll
    for (int i = 0; i < MFRAG; ++i) {
      const int am = wave * WR + i * 16 + nl;
      fa[i].q[0] = *(const uint4*)&As[am][hf * 8];
      fa[i].q[1] = *(const uint4*)&As[am][16 + hf * 8];
    }
    FragBF fb[4];
#pragma unroll
    for (int c = 0; c < 4; ++c) {
      const int bn = c * 16 + nl;
      fb[c].q[0] = *(const uint4*)&Bs[bn][hf * 8];
      fb[c].q[1] = *(const uint4*)&Bs[bn][16 + hf * 8];
    }
#pragma unroll
    for (int c = 0; c < 4; ++c)
#pragma unroll
      for (int i = 0; i < MFRAG; ++i)
        acc[i][c] = __builtin_amdgcn_wmma_f32_16x16x32_bf16(
            /*neg_a=*/false, fa[i].v, /*neg_b=*/false, fb[c].v,
            /*c_mod=*/(short)0, acc[i][c], /*reuse_a=*/false, /*reuse_b=*/false);
    __syncthreads();
  }

  // ---- epilogue: C/D layout -> lane nl = N, VGPR r -> M = hf*8 + r ----
#pragma unroll
  for (int c = 0; c < 4; ++c) {
    const int gn = tileN + c * 16 + nl;
    if (gn >= N) continue;
    const float bv = bias ? bias[gn] : 0.0f;
#pragma unroll
    for (int i = 0; i < MFRAG; ++i) {
#pragma unroll
      for (int r = 0; r < 8; ++r) {
        const int gm = tileM + wave * WR + i * 16 + hf * 8 + r;
        if (gm >= M) continue;
        float v = acc[i][c][r] + bv;
        if (act == 1) v = gelu_f(v);
        if (rowscale) v *= rowscale[gm];
        if (addsrc)   v += addsrc[(long)gm * ldc + gn];
        C[(long)gm * ldc + gn] = v;
      }
    }
  }
}

// ---------------------------------------------------------------------------
// Row LayerNorm (optionally fused GELU after the affine)
// ---------------------------------------------------------------------------
__global__ __launch_bounds__(256)
void ln_kernel(const float* __restrict__ x, const float* __restrict__ g,
               const float* __restrict__ b, float* __restrict__ y,
               int cols, int act) {
  const int row = blockIdx.x;
  const float* xr = x + (long)row * cols;
  float* yr = y + (long)row * cols;
  __shared__ float red[256];

  float s = 0.0f;
  for (int c = threadIdx.x; c < cols; c += 256) s += xr[c];
  red[threadIdx.x] = s; __syncthreads();
  for (int o = 128; o > 0; o >>= 1) {
    if (threadIdx.x < o) red[threadIdx.x] += red[threadIdx.x + o];
    __syncthreads();
  }
  const float mean = red[0] / (float)cols;
  __syncthreads();

  float v = 0.0f;
  for (int c = threadIdx.x; c < cols; c += 256) { float d = xr[c] - mean; v += d * d; }
  red[threadIdx.x] = v; __syncthreads();
  for (int o = 128; o > 0; o >>= 1) {
    if (threadIdx.x < o) red[threadIdx.x] += red[threadIdx.x + o];
    __syncthreads();
  }
  const float rstd = rsqrtf(red[0] / (float)cols + 1e-5f);

  for (int c = threadIdx.x; c < cols; c += 256) {
    float val = (xr[c] - mean) * rstd * g[c] + b[c];
    if (act) val = gelu_f(val);
    yr[c] = val;
  }
}

// ---------------------------------------------------------------------------
// padding_mask (bool bytes) -> float
// ---------------------------------------------------------------------------
__global__ void mask_f_kernel(const unsigned char* __restrict__ m, float* __restrict__ out, int n) {
  int i = blockIdx.x * blockDim.x + threadIdx.x;
  if (i < n) out[i] = m[i] ? 1.0f : 0.0f;
}

// ---------------------------------------------------------------------------
// gatt = sigmoid(expanded @ ga_W + ga_b)   (N = G = 8; one wave per output)
// ---------------------------------------------------------------------------
__global__ __launch_bounds__(256)
void ga_sigmoid_kernel(const float* __restrict__ x, const float* __restrict__ Wg,
                       const float* __restrict__ bg, float* __restrict__ out, int Kd) {
  const int row  = blockIdx.x;
  const int g    = threadIdx.x >> 5;     // 8 waves == G
  const int lane = threadIdx.x & 31;
  const float* xr = x + (long)row * Kd;
  float s = 0.0f;
  for (int k = lane; k < Kd; k += 32) s += xr[k] * Wg[k * Gc + g];
  for (int o = 16; o > 0; o >>= 1) s += __shfl_down(s, o, 32);
  if (lane == 0) {
    float v = s + bg[g];
    out[(long)row * Gc + g] = 1.0f / (1.0f + expf(-v));
  }
}

// ---------------------------------------------------------------------------
// In-place: softmax over K=32 within each (row, g) then scale by gatt*mf
// One wave per group (K == wave32 exactly).
// ---------------------------------------------------------------------------
__global__ __launch_bounds__(256)
void softmax_act_kernel(float* __restrict__ logits, const float* __restrict__ gatt,
                        const float* __restrict__ mf) {
  const int row  = blockIdx.x;
  const int g    = threadIdx.x >> 5;
  const int lane = threadIdx.x & 31;
  const long base = (long)row * (Gc * Kc) + (long)g * Kc;
  float v = logits[base + lane];
  float m = v;
  for (int o = 16; o > 0; o >>= 1) m = fmaxf(m, __shfl_xor(m, o, 32));
  float e = expf(v - m);
  float s = e;
  for (int o = 16; o > 0; o >>= 1) s += __shfl_xor(s, o, 32);
  const float scale = gatt[(long)row * Gc + g] * mf[row];
  logits[base + lane] = (e / s) * scale;
}

// ---------------------------------------------------------------------------
// act_sum[b,k] = sum_{t,g} act[b,tg,k]
// ---------------------------------------------------------------------------
__global__ __launch_bounds__(256)
void act_sum_kernel(const float* __restrict__ act, float* __restrict__ out) {
  const int b   = blockIdx.x;
  const int k   = threadIdx.x & 31;
  const int sub = threadIdx.x >> 5;     // 8 partials per k
  const int TG  = Tc * Gc;
  const float* ab = act + (long)b * TG * Kc;
  float s = 0.0f;
  for (int tg = sub; tg < TG; tg += 8) s += ab[(long)tg * Kc + k];
  __shared__ float red[256];
  red[threadIdx.x] = s; __syncthreads();
  if (sub == 0) {
    float t = 0.0f;
    for (int i = 0; i < 8; ++i) t += red[i * 32 + k];
    out[(long)b * Kc + k] = t;
  }
}

// ---------------------------------------------------------------------------
// res[b,k,:] = normalize( weighted[b,k,:] - act_sum[b,k]*centers[k,:] )
// One block per (b,k); blockDim == DG == 256.
// ---------------------------------------------------------------------------
__global__ __launch_bounds__(256)
void res_norm_kernel(const float* __restrict__ weighted, const float* __restrict__ act_sum,
                     const float* __restrict__ centers, float* __restrict__ res) {
  const int bk = blockIdx.x;            // b*K + k
  const int k  = bk & (Kc - 1);
  const int d  = threadIdx.x;
  const float asum = act_sum[bk];
  const float v = weighted[(long)bk * DGc + d] - asum * centers[(long)k * DGc + d];
  __shared__ float red[256];
  red[d] = v * v; __syncthreads();
  for (int o = 128; o > 0; o >>= 1) {
    if (d < o) red[d] += red[d + o];
    __syncthreads();
  }
  const float inv = 1.0f / fmaxf(sqrtf(red[0]), 1e-12f);
  res[(long)bk * DGc + d] = v * inv;
}

// ---------------------------------------------------------------------------
// out[n] = b[n] + sum_k x[k]*W[k,n]   (1 x K row-vector GEMV; coalesced over n)
// ---------------------------------------------------------------------------
__global__ void gemv_rowvec_kernel(const float* __restrict__ x, const float* __restrict__ W,
                                   const float* __restrict__ b, float* __restrict__ out,
                                   int Kd, int N) {
  const int n = blockIdx.x * blockDim.x + threadIdx.x;
  if (n >= N) return;
  float s = b[n];
  for (int k = 0; k < Kd; ++k) s += x[k] * W[(long)k * N + n];
  out[n] = s;
}

// ---------------------------------------------------------------------------
// scores[b,h,t] = dot(q[h,:], k[b,t,h,:]) / sqrt(DH), masked. One wave each.
// ---------------------------------------------------------------------------
__global__ __launch_bounds__(128)
void attn_scores_kernel(const float* __restrict__ q, const float* __restrict__ kmat,
                        const unsigned char* __restrict__ mask, float* __restrict__ scores) {
  const int idx  = blockIdx.x * 4 + (threadIdx.x >> 5);
  const int lane = threadIdx.x & 31;
  const int t  = idx % Tc;
  const int bh = idx / Tc;
  const int h  = bh % Hc;
  const int b  = bh / Hc;
  if (b >= Bc) return;
  const float* kr = kmat + ((long)(b * Tc + t) * Dc) + h * DHc;
  const float* qr = q + h * DHc;
  float s = 0.0f;
  for (int d = lane; d < DHc; d += 32) s += qr[d] * kr[d];
  for (int o = 16; o > 0; o >>= 1) s += __shfl_down(s, o, 32);
  if (lane == 0) {
    float val = s * rsqrtf((float)DHc);
    if (!mask[b * Tc + t]) val = -1e9f;
    scores[(long)bh * Tc + t] = val;
  }
}

// ---------------------------------------------------------------------------
// Generic row softmax (block per row)
// ---------------------------------------------------------------------------
__global__ __launch_bounds__(256)
void softmax_rows_kernel(const float* __restrict__ x, float* __restrict__ y, int cols) {
  const int row = blockIdx.x;
  const float* xr = x + (long)row * cols;
  float* yr = y + (long)row * cols;
  __shared__ float red[256];

  float m = -3.4e38f;
  for (int c = threadIdx.x; c < cols; c += 256) m = fmaxf(m, xr[c]);
  red[threadIdx.x] = m; __syncthreads();
  for (int o = 128; o > 0; o >>= 1) {
    if (threadIdx.x < o) red[threadIdx.x] = fmaxf(red[threadIdx.x], red[threadIdx.x + o]);
    __syncthreads();
  }
  m = red[0]; __syncthreads();

  float s = 0.0f;
  for (int c = threadIdx.x; c < cols; c += 256) s += expf(xr[c] - m);
  red[threadIdx.x] = s; __syncthreads();
  for (int o = 128; o > 0; o >>= 1) {
    if (threadIdx.x < o) red[threadIdx.x] += red[threadIdx.x + o];
    __syncthreads();
  }
  const float inv = 1.0f / red[0];
  for (int c = threadIdx.x; c < cols; c += 256) yr[c] = expf(xr[c] - m) * inv;
}

// ---------------------------------------------------------------------------
// ctx[b, h*DH+d] = sum_t attn[b,h,t] * v[b,t,h*DH+d]. One block per (b,h).
// ---------------------------------------------------------------------------
__global__ __launch_bounds__(128)
void attn_ctx_kernel(const float* __restrict__ attn, const float* __restrict__ vmat,
                     float* __restrict__ ctx) {
  const int bh = blockIdx.x;
  const int h = bh % Hc, b = bh / Hc;
  const int d = threadIdx.x;          // DH == 128
  const float* ar = attn + (long)bh * Tc;
  const float* vb = vmat + (long)b * Tc * Dc + h * DHc + d;
  float s = 0.0f;
  for (int t = 0; t < Tc; ++t) s += ar[t] * vb[(long)t * Dc];
  ctx[(long)b * Dc + h * DHc + d] = s;
}

// ---------------------------------------------------------------------------
// concat along last dim
// ---------------------------------------------------------------------------
__global__ void concat_kernel(const float* __restrict__ a, const float* __restrict__ b,
                              float* __restrict__ out, int rows, int ca, int cb) {
  const int i = blockIdx.x * blockDim.x + threadIdx.x;
  const int cols = ca + cb;
  if (i >= rows * cols) return;
  const int r = i / cols, c = i % cols;
  out[i] = (c < ca) ? a[(long)r * ca + c] : b[(long)r * cb + (c - ca)];
}

// ---------------------------------------------------------------------------
// Host-side launcher: pick 128-row tile for tall GEMMs, 64-row for batch-32
// ---------------------------------------------------------------------------
static inline void launch_gemm(hipStream_t s, const float* A, const float* W,
                               const float* bias, const float* rowscale,
                               const float* addsrc, float* C,
                               int M, int N, int Kd, int lda, int ldw, int ldc,
                               long sA, long sW, long sC, int batch, int transA, int act) {
  if (M >= 128) {
    dim3 grid((N + TN - 1) / TN, (M + 127) / 128, batch);
    gemm_wmma_kernel<2><<<grid, GEMM_THREADS, 0, s>>>(A, W, bias, rowscale, addsrc, C,
                                                      M, N, Kd, lda, ldw, ldc, sA, sW, sC,
                                                      transA, act);
  } else {
    dim3 grid((N + TN - 1) / TN, (M + 63) / 64, batch);
    gemm_wmma_kernel<1><<<grid, GEMM_THREADS, 0, s>>>(A, W, bias, rowscale, addsrc, C,
                                                      M, N, Kd, lda, ldw, ldc, sA, sW, sC,
                                                      transA, act);
  }
}

extern "C" void kernel_launch(void* const* d_in, const int* in_sizes, int n_in,
                              void* d_out, int out_size, void* d_ws, size_t ws_size,
                              hipStream_t stream) {
  (void)in_sizes; (void)n_in; (void)out_size; (void)ws_size;

  const float* vf        = (const float*)d_in[0];
  const unsigned char* pmask = (const unsigned char*)d_in[1];
  const float* exp_ln_g  = (const float*)d_in[2];
  const float* exp_ln_b  = (const float*)d_in[3];
  const float* exp_W     = (const float*)d_in[4];
  const float* exp_b     = (const float*)d_in[5];
  const float* ga_W      = (const float*)d_in[6];
  const float* ga_b      = (const float*)d_in[7];
  const float* ca_W      = (const float*)d_in[8];
  const float* ca_b      = (const float*)d_in[9];
  const float* centers   = (const float*)d_in[10];
  const float* vlad_ln1_g = (const float*)d_in[11];
  const float* vlad_ln1_b = (const float*)d_in[12];
  const float* vlad_W    = (const float*)d_in[13];
  const float* vlad_b    = (const float*)d_in[14];
  const float* vlad_ln2_g = (const float*)d_in[15];
  const float* vlad_ln2_b = (const float*)d_in[16];
  const float* pool_query = (const float*)d_in[17];
  const float* Wq        = (const float*)d_in[18];
  const float* bq        = (const float*)d_in[19];
  const float* Wk        = (const float*)d_in[20];
  const float* bk        = (const float*)d_in[21];
  const float* Wv        = (const float*)d_in[22];
  const float* bv        = (const float*)d_in[23];
  const float* Wo        = (const float*)d_in[24];
  const float* bo        = (const float*)d_in[25];
  const float* pool_ln_g = (const float*)d_in[26];
  const float* pool_ln_b = (const float*)d_in[27];
  const float* ap_ln1_g  = (const float*)d_in[28];
  const float* ap_ln1_b  = (const float*)d_in[29];
  const float* ap_W      = (const float*)d_in[30];
  const float* ap_b      = (const float*)d_in[31];
  const float* ap_ln2_g  = (const float*)d_in[32];
  const float* ap_ln2_b  = (const float*)d_in[33];
  const float* f_W1      = (const float*)d_in[34];
  const float* f_b1      = (const float*)d_in[35];
  const float* f_ln1_g   = (const float*)d_in[36];
  const float* f_ln1_b   = (const float*)d_in[37];
  const float* f_W2      = (const float*)d_in[38];
  const float* f_b2      = (const float*)d_in[39];
  const float* f_ln2_g   = (const float*)d_in[40];
  const float* f_ln2_b   = (const float*)d_in[41];
  const float* in_W      = (const float*)d_in[42];
  const float* in_b      = (const float*)d_in[43];
  const float* in_ln_g   = (const float*)d_in[44];
  const float* in_ln_b   = (const float*)d_in[45];
  const float* blk_norm_g = (const float*)d_in[46];
  const float* blk_norm_b = (const float*)d_in[47];
  const float* blk_W1    = (const float*)d_in[48];
  const float* blk_b1    = (const float*)d_in[49];
  const float* blk_W2    = (const float*)d_in[50];
  const float* blk_b2    = (const float*)d_in[51];
  const float* out_ln_g  = (const float*)d_in[52];
  const float* out_ln_b  = (const float*)d_in[53];
  const float* out_W     = (const float*)d_in[54];
  const float* out_b     = (const float*)d_in[55];

  float* outp = (float*)d_out;

  // ---- workspace bump allocator (256B aligned) ----
  char* wsp = (char*)d_ws;
  size_t woff = 0;
  auto alloc = [&](size_t nfloats) -> float* {
    float* p = (float*)(wsp + woff);
    woff += ((nfloats * sizeof(float) + 255) / 256) * 256;
    return p;
  };

  float* mf       = alloc(BT);
  float* ln_vf    = alloc((size_t)BT * Dc);
  float* expanded = alloc((size_t)BT * Ec);
  float* gatt     = alloc((size_t)BT * Gc);
  float* actb     = alloc((size_t)BT * Gc * Kc);      // ca logits -> act (in place)
  float* act_sum  = alloc((size_t)Bc * Kc);
  float* weighted = alloc((size_t)Bc * Kc * DGc);
  float* res      = alloc((size_t)Bc * Kc * DGc);
  float* res_ln   = alloc((size_t)Bc * Kc * DGc);
  float* vlad_pre = alloc((size_t)Bc * HIDc);
  float* vlad     = alloc((size_t)Bc * HIDc);
  float* qvec     = alloc(Dc);
  float* kmat     = alloc((size_t)BT * Dc);
  float* vmat     = alloc((size_t)BT * Dc);
  float* scores   = alloc((size_t)Bc * Hc * Tc);
  float* attn     = alloc((size_t)Bc * Hc * Tc);
  float* ctx      = alloc((size_t)Bc * Dc);
  float* pooled_pre = alloc((size_t)Bc * Dc);
  float* pooled   = alloc((size_t)Bc * Dc);
  float* t1       = alloc((size_t)Bc * Dc);
  float* t2       = alloc((size_t)Bc * HIDc);
  float* astate   = alloc((size_t)Bc * HIDc);
  float* catb     = alloc((size_t)Bc * 2 * HIDc);
  float* x1       = alloc((size_t)Bc * HIDc);
  float* xg       = alloc((size_t)Bc * HIDc);
  float* x2       = alloc((size_t)Bc * Dc);
  float* xx       = alloc((size_t)Bc * Dc);
  float* h_pre    = alloc((size_t)Bc * HIDc);
  float* hA       = alloc((size_t)Bc * HIDc);
  float* hB       = alloc((size_t)Bc * HIDc);
  float* r1       = alloc((size_t)Bc * HIDc);
  float* r2       = alloc((size_t)Bc * 2 * HIDc);
  float* hln      = alloc((size_t)Bc * HIDc);

  // ============================ NeXtVLAD branch ============================
  mask_f_kernel<<<(BT + 255) / 256, 256, 0, stream>>>(pmask, mf, BT);
  ln_kernel<<<BT, 256, 0, stream>>>(vf, exp_ln_g, exp_ln_b, ln_vf, Dc, 0);

  // expanded = (LN(vf) @ exp_W + exp_b) * mf
  launch_gemm(stream, ln_vf, exp_W, exp_b, mf, nullptr, expanded,
              BT, Ec, Dc, Dc, Ec, Ec, 0, 0, 0, 1, 0, 0);

  // gatt = sigmoid(expanded @ ga_W + ga_b)
  ga_sigmoid_kernel<<<BT, 256, 0, stream>>>(expanded, ga_W, ga_b, gatt, Ec);

  // ca logits = expanded @ ca_W + ca_b
  launch_gemm(stream, expanded, ca_W, ca_b, nullptr, nullptr, actb,
              BT, Gc * Kc, Ec, Ec, Gc * Kc, Gc * Kc, 0, 0, 0, 1, 0, 0);

  // act = softmax_K(logits) * gatt * mf   (in place)
  softmax_act_kernel<<<BT, 256, 0, stream>>>(actb, gatt, mf);

  // act_sum[b,k]
  act_sum_kernel<<<Bc, 256, 0, stream>>>(actb, act_sum);

  // weighted[b] = act_b^T (32 x 4096) @ grouped_b (4096 x 256)   -- batched WMMA
  launch_gemm(stream, actb, expanded, nullptr, nullptr, nullptr, weighted,
              Kc, DGc, Tc * Gc, /*lda=*/Kc, /*ldw=*/DGc, /*ldc=*/DGc,
              /*sA=*/(long)Tc * Gc * Kc, /*sW=*/(long)Tc * Gc * DGc,
              /*sC=*/(long)Kc * DGc, /*batch=*/Bc, /*transA=*/1, 0);

  // residual to centers + L2 normalize per (b,k)
  res_norm_kernel<<<Bc * Kc, 256, 0, stream>>>(weighted, act_sum, centers, res);

  // vlad = LN2( GELU( LN1(res) @ vlad_W + vlad_b ) )
  ln_kernel<<<Bc, 256, 0, stream>>>(res, vlad_ln1_g, vlad_ln1_b, res_ln, Kc * DGc, 0);
  launch_gemm(stream, res_ln, vlad_W, vlad_b, nullptr, nullptr, vlad_pre,
              Bc, HIDc, Kc * DGc, Kc * DGc, HIDc, HIDc, 0, 0, 0, 1, 0, 1);
  ln_kernel<<<Bc, 256, 0, stream>>>(vlad_pre, vlad_ln2_g, vlad_ln2_b, vlad, HIDc, 0);

  // ========================= attention-pool branch =========================
  gemv_rowvec_kernel<<<(Dc + 255) / 256, 256, 0, stream>>>(pool_query, Wq, bq, qvec, Dc, Dc);
  launch_gemm(stream, vf, Wk, bk, nullptr, nullptr, kmat,
              BT, Dc, Dc, Dc, Dc, Dc, 0, 0, 0, 1, 0, 0);
  launch_gemm(stream, vf, Wv, bv, nullptr, nullptr, vmat,
              BT, Dc, Dc, Dc, Dc, Dc, 0, 0, 0, 1, 0, 0);

  attn_scores_kernel<<<(Bc * Hc * Tc) / 4, 128, 0, stream>>>(qvec, kmat, pmask, scores);
  softmax_rows_kernel<<<Bc * Hc, 256, 0, stream>>>(scores, attn, Tc);
  attn_ctx_kernel<<<Bc * Hc, 128, 0, stream>>>(attn, vmat, ctx);

  launch_gemm(stream, ctx, Wo, bo, nullptr, nullptr, pooled_pre,
              Bc, Dc, Dc, Dc, Dc, Dc, 0, 0, 0, 1, 0, 0);
  ln_kernel<<<Bc, 256, 0, stream>>>(pooled_pre, pool_ln_g, pool_ln_b, pooled, Dc, 0);

  ln_kernel<<<Bc, 256, 0, stream>>>(pooled, ap_ln1_g, ap_ln1_b, t1, Dc, 0);
  launch_gemm(stream, t1, ap_W, ap_b, nullptr, nullptr, t2,
              Bc, HIDc, Dc, Dc, HIDc, HIDc, 0, 0, 0, 1, 0, 1);   // GELU epilogue
  ln_kernel<<<Bc, 256, 0, stream>>>(t2, ap_ln2_g, ap_ln2_b, astate, HIDc, 0);

  // ================================ fuse ===================================
  concat_kernel<<<(Bc * 2 * HIDc + 255) / 256, 256, 0, stream>>>(vlad, astate, catb,
                                                                 Bc, HIDc, HIDc);
  launch_gemm(stream, catb, f_W1, f_b1, nullptr, nullptr, x1,
              Bc, HIDc, 2 * HIDc, 2 * HIDc, HIDc, HIDc, 0, 0, 0, 1, 0, 0);
  ln_kernel<<<Bc, 256, 0, stream>>>(x1, f_ln1_g, f_ln1_b, xg, HIDc, /*gelu=*/1);
  launch_gemm(stream, xg, f_W2, f_b2, nullptr, nullptr, x2,
              Bc, Dc, HIDc, HIDc, Dc, Dc, 0, 0, 0, 1, 0, 0);
  ln_kernel<<<Bc, 256, 0, stream>>>(x2, f_ln2_g, f_ln2_b, xx, Dc, 0);

  // ===================== deep residual MLP classifier ======================
  launch_gemm(stream, xx, in_W, in_b, nullptr, nullptr, h_pre,
              Bc, HIDc, Dc, Dc, HIDc, HIDc, 0, 0, 0, 1, 0, 0);
  ln_kernel<<<Bc, 256, 0, stream>>>(h_pre, in_ln_g, in_ln_b, hA, HIDc, 0);

  float* hcur = hA;
  float* hnxt = hB;
  for (int i = 0; i < NBc; ++i) {
    ln_kernel<<<Bc, 256, 0, stream>>>(hcur, blk_norm_g + (size_t)i * HIDc,
                                      blk_norm_b + (size_t)i * HIDc, r1, HIDc, 0);
    launch_gemm(stream, r1, blk_W1 + (size_t)i * HIDc * 2 * HIDc,
                blk_b1 + (size_t)i * 2 * HIDc, nullptr, nullptr, r2,
                Bc, 2 * HIDc, HIDc, HIDc, 2 * HIDc, 2 * HIDc, 0, 0, 0, 1, 0, 1); // GELU
    launch_gemm(stream, r2, blk_W2 + (size_t)i * 2 * HIDc * HIDc,
                blk_b2 + (size_t)i * HIDc, nullptr, /*addsrc=*/hcur, hnxt,
                Bc, HIDc, 2 * HIDc, 2 * HIDc, HIDc, HIDc, 0, 0, 0, 1, 0, 0);
    float* tmp = hcur; hcur = hnxt; hnxt = tmp;
  }

  ln_kernel<<<Bc, 256, 0, stream>>>(hcur, out_ln_g, out_ln_b, hln, HIDc, 0);
  launch_gemm(stream, hln, out_W, out_b, nullptr, nullptr, outp,
              Bc, NCc, HIDc, HIDc, NCc, NCc, 0, 0, 0, 1, 0, 0);
}
